// LinformerAttention_31945966748101
// MI455X (gfx1250) — matrix operand order repaired and checked by reading, not compiled, not run on hardware
//
#include <hip/hip_runtime.h>
#include <hip/hip_bf16.h>
#include <cstdint>
#include <cstddef>

// ---------------------------------------------------------------------------
// CDNA5 (gfx1250) Linformer attention, bf16 WMMA pipeline.
// ---------------------------------------------------------------------------

typedef __attribute__((ext_vector_type(16))) __bf16 v16bf;
typedef __attribute__((ext_vector_type(8)))  float  v8f;

union FragB16 {
    v16bf    v;
    unsigned u[8];
};

union U4 {
    uint4    q;
    unsigned u[4];
    __bf16   h[8];
};

__device__ __forceinline__ v8f v8f_zero() {
    v8f z = {0.f, 0.f, 0.f, 0.f, 0.f, 0.f, 0.f, 0.f};
    return z;
}

// A-fragment K index for 16-bit 16x16x32 WMMA:
// lanes 0-15: V0..V3 -> K 0..7, V4..V7 -> K 16..23
// lanes 16-31: shift K by +8
__device__ __forceinline__ int a_k0(int v, int half) {
    return ((v < 4) ? 2 * v : 8 + 2 * v) + half * 8;
}

// ---------------------------------------------------------------------------
// Elementwise / transpose conversion kernels (fp32 -> bf16)
// ---------------------------------------------------------------------------
__global__ void f32_to_bf16_kernel(const float* __restrict__ in,
                                   __bf16* __restrict__ out, size_t n) {
    size_t i = (size_t)blockIdx.x * blockDim.x + threadIdx.x;
    size_t stride = (size_t)gridDim.x * blockDim.x;
    for (; i < n; i += stride) out[i] = (__bf16)in[i];
}

// out[c*rows + r] = in[r*cols + c]   (projT[kproj][t] = proj[t][kproj])
__global__ void transpose_f32_to_bf16_kernel(const float* __restrict__ in,
                                             __bf16* __restrict__ out,
                                             int rows, int cols) {
    size_t n = (size_t)rows * cols;
    size_t i = (size_t)blockIdx.x * blockDim.x + threadIdx.x;
    size_t stride = (size_t)gridDim.x * blockDim.x;
    for (; i < n; i += stride) {
        size_t r = i / (size_t)cols;
        size_t c = i % (size_t)cols;
        out[c * (size_t)rows + r] = (__bf16)in[i];
    }
}

// ---------------------------------------------------------------------------
// FAST contiguous bf16 GEMM: C[MxN] = A[MxK] * B[KxN], all row-major, dims
// divisible by tile (M%128==0, N%128==0, K%32==0), 16B-aligned pointers.
// BM=BN=128, BK=32; 256 threads = 8 waves; each wave computes 32x64
// (2x4 grid of 16x16 WMMA tiles -> 8 WMMAs per K-chunk per wave).
// Global traffic: b128 loads; next-chunk global_prefetch_b8.
// ---------------------------------------------------------------------------
template <bool OUT_BF16>
__global__ void gemm_bf16_wmma_fast(const __bf16* __restrict__ A,
                                    const __bf16* __restrict__ B,
                                    void* __restrict__ C,
                                    int M, int N, int K) {
    // stride 40 elems = 80B: keeps [r][8m] stores 16B aligned; 20-bank row
    // pitch keeps 16-row fragment reads conflict-free (gcd(20,64)=4).
    __shared__ __bf16 sA[128][40];
    __shared__ __bf16 sBt[128][40];   // B tile transposed: [n][k]

    const int tid  = threadIdx.x;
    const int lane = tid & 31;
    const int wid  = tid >> 5;
    const int wm   = wid >> 1;        // 0..3 -> row block of 32
    const int wn   = wid & 1;         // 0..1 -> col block of 64
    const int half = lane >> 4;
    const int l15  = lane & 15;

    const int tileN = blockIdx.x * 128;
    const int tileM = blockIdx.y * 128;

    v8f acc[2][4];
#pragma unroll
    for (int i = 0; i < 2; ++i)
#pragma unroll
        for (int j = 0; j < 4; ++j) acc[i][j] = v8f_zero();

    for (int k0 = 0; k0 < K; k0 += 32) {
        __syncthreads();
        // stage A tile 128x32: 512 x uint4, 2 per thread
#pragma unroll
        for (int t = 0; t < 2; ++t) {
            int idx = tid + t * 256;            // 0..511
            int r = idx >> 2, kk = (idx & 3) * 8;
            const __bf16* src = &A[(size_t)(tileM + r) * K + k0 + kk];
            U4 val;
            val.q = *(const uint4*)src;
            *(uint4*)&sA[r][kk] = val.q;
            if (k0 + 32 < K) __builtin_prefetch(src + 32, 0, 1);
        }
        // stage B tile 32x128 transposed: 512 x uint4 loads, scatter to sBt
#pragma unroll
        for (int t = 0; t < 2; ++t) {
            int idx = tid + t * 256;            // 0..511
            int kk = idx >> 4, nblk = (idx & 15) * 8;
            const __bf16* src = &B[(size_t)(k0 + kk) * N + tileN + nblk];
            U4 val;
            val.q = *(const uint4*)src;
#pragma unroll
            for (int e = 0; e < 8; ++e) sBt[nblk + e][kk] = val.h[e];
            if (k0 + 32 < K) __builtin_prefetch(src + (size_t)32 * N, 0, 1);
        }
        __syncthreads();

        FragB16 af[2], bf[4];
#pragma unroll
        for (int i = 0; i < 2; ++i) {
            int row = wm * 32 + i * 16 + l15;
#pragma unroll
            for (int v = 0; v < 8; ++v)
                af[i].u[v] = *(const unsigned*)&sA[row][a_k0(v, half)];
        }
#pragma unroll
        for (int j = 0; j < 4; ++j) {
            int col = wn * 64 + j * 16 + l15;
#pragma unroll
            for (int v = 0; v < 8; ++v)
                bf[j].u[v] = *(const unsigned*)&sBt[col][a_k0(v, half)];
        }
#pragma unroll
        for (int i = 0; i < 2; ++i)
#pragma unroll
            for (int j = 0; j < 4; ++j)
                acc[i][j] = __builtin_amdgcn_wmma_f32_16x16x32_bf16(
                    false, af[i].v, false, bf[j].v, (short)0, acc[i][j],
                    false, false);
    }

#pragma unroll
    for (int i = 0; i < 2; ++i)
#pragma unroll
        for (int j = 0; j < 4; ++j)
#pragma unroll
            for (int e = 0; e < 8; ++e) {
                int row = tileM + wm * 32 + i * 16 + e + 8 * half;
                int col = tileN + wn * 64 + j * 16 + l15;
                size_t off = (size_t)row * N + col;
                if constexpr (OUT_BF16)
                    ((__bf16*)C)[off] = (__bf16)acc[i][j][e];
                else
                    ((float*)C)[off] = acc[i][j][e];
            }
}

// ---------------------------------------------------------------------------
// Generic strided batched bf16 GEMM (element-granular loads, bounds-guarded):
//   A: (m,k) at A[offA + m*rsA + k*csA];  B: (k,n) at B[offB + k*rsB + n*csB]
//   C: (m,n) at C[offC + m*rsC + n*csC]; batch z: off = (z/bdiv)*s0+(z%bdiv)*s1
// BM=128, BN=64, BK=32; 8 waves, each a 32x32 patch (2x2 WMMA tiles).
// Used for the low-rank K/V projections whose B operand is the interleaved
// qkv view (column stride 48) -- not vectorizable anyway.
// ---------------------------------------------------------------------------
template <bool OUT_BF16>
__global__ void gemm_bf16_wmma(const __bf16* __restrict__ A,
                               const __bf16* __restrict__ B,
                               void* __restrict__ C,
                               int M, int N, int K,
                               long long rsA, long long csA,
                               long long rsB, long long csB,
                               long long rsC, long long csC,
                               int bdiv,
                               long long baA0, long long baA1,
                               long long baB0, long long baB1,
                               long long baC0, long long baC1) {
    __shared__ __bf16 sA[128][34];
    __shared__ __bf16 sBt[64][34];

    const int tid  = threadIdx.x;
    const int lane = tid & 31;
    const int wid  = tid >> 5;
    const int wm   = wid >> 1;
    const int wn   = wid & 1;
    const int half = lane >> 4;
    const int l15  = lane & 15;

    const int tileN = blockIdx.x * 64;
    const int tileM = blockIdx.y * 128;
    const int z     = blockIdx.z;
    const long long offA = (long long)(z / bdiv) * baA0 + (long long)(z % bdiv) * baA1;
    const long long offB = (long long)(z / bdiv) * baB0 + (long long)(z % bdiv) * baB1;
    const long long offC = (long long)(z / bdiv) * baC0 + (long long)(z % bdiv) * baC1;

    v8f acc[2][2];
#pragma unroll
    for (int i = 0; i < 2; ++i)
#pragma unroll
        for (int j = 0; j < 2; ++j) acc[i][j] = v8f_zero();

    for (int k0 = 0; k0 < K; k0 += 32) {
        __syncthreads();
        for (int idx = tid; idx < 128 * 32; idx += 256) {
            int r = idx >> 5, kk = idx & 31;
            int gm = tileM + r, gk = k0 + kk;
            __bf16 val = (__bf16)0.0f;
            if (gm < M && gk < K)
                val = A[offA + (long long)gm * rsA + (long long)gk * csA];
            sA[r][kk] = val;
        }
        for (int idx = tid; idx < 64 * 32; idx += 256) {
            int n = idx & 63, kk = idx >> 6;
            int gn = tileN + n, gk = k0 + kk;
            __bf16 val = (__bf16)0.0f;
            if (gn < N && gk < K)
                val = B[offB + (long long)gk * rsB + (long long)gn * csB];
            sBt[n][kk] = val;
        }
        __syncthreads();

        FragB16 af[2], bf[2];
#pragma unroll
        for (int i = 0; i < 2; ++i) {
            int row = wm * 32 + i * 16 + l15;
#pragma unroll
            for (int v = 0; v < 8; ++v)
                af[i].u[v] = *(const unsigned*)&sA[row][a_k0(v, half)];
        }
#pragma unroll
        for (int j = 0; j < 2; ++j) {
            int col = wn * 32 + j * 16 + l15;
#pragma unroll
            for (int v = 0; v < 8; ++v)
                bf[j].u[v] = *(const unsigned*)&sBt[col][a_k0(v, half)];
        }
#pragma unroll
        for (int i = 0; i < 2; ++i)
#pragma unroll
            for (int j = 0; j < 2; ++j)
                acc[i][j] = __builtin_amdgcn_wmma_f32_16x16x32_bf16(
                    false, af[i].v, false, bf[j].v, (short)0, acc[i][j],
                    false, false);
    }

#pragma unroll
    for (int i = 0; i < 2; ++i)
#pragma unroll
        for (int j = 0; j < 2; ++j)
#pragma unroll
            for (int e = 0; e < 8; ++e) {
                int row = tileM + wm * 32 + i * 16 + e + 8 * half;
                int col = tileN + wn * 32 + j * 16 + l15;
                if (row < M && col < N) {
                    long long off = offC + (long long)row * rsC + (long long)col * csC;
                    if constexpr (OUT_BF16)
                        ((__bf16*)C)[off] = (__bf16)acc[i][j][e];
                    else
                        ((float*)C)[off] = acc[i][j][e];
                }
            }
}

// ---------------------------------------------------------------------------
// Fused attention: per block = one (b,h) and 128 query rows.
// KPROJ=256 iterated in 4 chunks of 64 with online (flash) softmax.
//   q(t,d)  = qkv[(b*T+t)*3072 + d*48 + h]            (bf16, interleaved)
//   kp/vp   = [B*H][256][64] bf16
//   out     = [B*T][1024] bf16 at column h*64+d
// ---------------------------------------------------------------------------
__global__ void linformer_attn_kernel(const __bf16* __restrict__ qkv,
                                      const __bf16* __restrict__ kp,
                                      const __bf16* __restrict__ vp,
                                      __bf16* __restrict__ outb) {
    constexpr int T = 4096;
    constexpr int QKVLD = 3072;

    __shared__ __bf16 sQ[128][66];     // query tile, [t][d]
    __shared__ __bf16 sKP[64][72];     // K' chunk,  [k'][d]  (== B^T for scores)
    __shared__ __bf16 sVT[64][72];     // V chunk^T, [d][k']  (== B^T for P@V)
    __shared__ __bf16 sP[8][16 * 64];  // per-wave P re-layout buffer

    const int tid  = threadIdx.x;
    const int lane = tid & 31;
    const int wv   = tid >> 5;         // 8 waves, 16 rows each
    const int half = lane >> 4;
    const int l15  = lane & 15;
    const int bh   = blockIdx.y;       // b*16 + h
    const int b    = bh >> 4;
    const int h    = bh & 15;
    const int t0   = blockIdx.x * 128;

    for (int idx = tid; idx < 128 * 64; idx += 256) {
        int r = idx >> 6, d = idx & 63;
        sQ[r][d] = qkv[(size_t)(b * T + t0 + r) * QKVLD + d * 48 + h];
    }

    v8f o_acc[4];
#pragma unroll
    for (int nd = 0; nd < 4; ++nd) o_acc[nd] = v8f_zero();
    float m[8], s[8];
#pragma unroll
    for (int i = 0; i < 8; ++i) { m[i] = -1e30f; s[i] = 0.f; }

    const size_t kvbase = (size_t)bh * 256 * 64;

    for (int c = 0; c < 4; ++c) {
        const int kc = c * 64;
        __syncthreads();
        // stage 64x64 kp chunk (vectorized) + vp chunk (transposed scatter)
#pragma unroll
        for (int t = 0; t < 2; ++t) {
            int idx = tid + t * 256;               // 0..511
            int i = idx >> 3, dblk = (idx & 7) * 8;
            const __bf16* ksrc = &kp[kvbase + (size_t)(kc + i) * 64 + dblk];
            const __bf16* vsrc = &vp[kvbase + (size_t)(kc + i) * 64 + dblk];
            U4 kv, vv;
            kv.q = *(const uint4*)ksrc;
            vv.q = *(const uint4*)vsrc;
            *(uint4*)&sKP[i][dblk] = kv.q;         // stride 72 -> 16B aligned
#pragma unroll
            for (int e = 0; e < 8; ++e) sVT[dblk + e][i] = vv.h[e];
        }
        __syncthreads();

        // ----- scores S = Q * KP^T for this chunk (16 rows x 64 cols / wave)
        v8f s_acc[4];
#pragma unroll
        for (int nb = 0; nb < 4; ++nb) s_acc[nb] = v8f_zero();
#pragma unroll
        for (int ks = 0; ks < 2; ++ks) {
            FragB16 qa;
            int qrow = wv * 16 + l15;
#pragma unroll
            for (int v = 0; v < 8; ++v)
                qa.u[v] = *(const unsigned*)&sQ[qrow][ks * 32 + a_k0(v, half)];
#pragma unroll
            for (int nb = 0; nb < 4; ++nb) {
                FragB16 bb;
                int col = nb * 16 + l15;
#pragma unroll
                for (int v = 0; v < 8; ++v)
                    bb.u[v] = *(const unsigned*)&sKP[col][ks * 32 + a_k0(v, half)];
                s_acc[nb] = __builtin_amdgcn_wmma_f32_16x16x32_bf16(
                    false, qa.v, false, bb.v, (short)0, s_acc[nb], false, false);
            }
        }

        // ----- scale + online softmax update (row = i + 8*half per lane)
#pragma unroll
        for (int nb = 0; nb < 4; ++nb)
#pragma unroll
            for (int i = 0; i < 8; ++i) s_acc[nb][i] *= 0.125f;  // DH^-0.5

#pragma unroll
        for (int i = 0; i < 8; ++i) {
            float cm = fmaxf(fmaxf(s_acc[0][i], s_acc[1][i]),
                             fmaxf(s_acc[2][i], s_acc[3][i]));
            for (int off = 1; off < 16; off <<= 1)
                cm = fmaxf(cm, __shfl_xor(cm, off, 32));
            float mnew = fmaxf(m[i], cm);
            float corr = __expf(m[i] - mnew);
            float rsum = 0.f;
#pragma unroll
            for (int nb = 0; nb < 4; ++nb) {
                float p = __expf(s_acc[nb][i] - mnew);
                s_acc[nb][i] = p;
                rsum += p;
            }
            for (int off = 1; off < 16; off <<= 1)
                rsum += __shfl_xor(rsum, off, 32);
            s[i] = s[i] * corr + rsum;
            m[i] = mnew;
#pragma unroll
            for (int nd = 0; nd < 4; ++nd) o_acc[nd][i] *= corr;
        }

        // ----- spill P (C-layout) to per-wave LDS, reload as A-fragments
#pragma unroll
        for (int nb = 0; nb < 4; ++nb)
#pragma unroll
            for (int i = 0; i < 8; ++i) {
                int row = i + 8 * half;
                sP[wv][row * 64 + nb * 16 + l15] = (__bf16)s_acc[nb][i];
            }
        // same-wave LDS RAW: LDS ops are in-order per wave

        // ----- O += P * V   (B^T from sVT)
#pragma unroll
        for (int ks = 0; ks < 2; ++ks) {
            FragB16 pa;
#pragma unroll
            for (int v = 0; v < 8; ++v)
                pa.u[v] = *(const unsigned*)&sP[wv][l15 * 64 + ks * 32 + a_k0(v, half)];
#pragma unroll
            for (int nd = 0; nd < 4; ++nd) {
                FragB16 vb;
                int col = nd * 16 + l15;
#pragma unroll
                for (int v = 0; v < 8; ++v)
                    vb.u[v] = *(const unsigned*)&sVT[col][ks * 32 + a_k0(v, half)];
                o_acc[nd] = __builtin_amdgcn_wmma_f32_16x16x32_bf16(
                    false, pa.v, false, vb.v, (short)0, o_acc[nd], false, false);
            }
        }
    }

    // ----- epilogue: normalize and scatter to [b,t, h*64+d]
#pragma unroll
    for (int nd = 0; nd < 4; ++nd)
#pragma unroll
        for (int i = 0; i < 8; ++i) {
            int row = t0 + wv * 16 + i + 8 * half;
            int col = nd * 16 + l15;
            float val = o_acc[nd][i] / s[i];
            outb[(size_t)(b * T + row) * 1024 + h * 64 + col] = (__bf16)val;
        }
}

// ---------------------------------------------------------------------------
// Host launcher
// ---------------------------------------------------------------------------
extern "C" void kernel_launch(void* const* d_in, const int* in_sizes, int n_in,
                              void* d_out, int out_size, void* d_ws, size_t ws_size,
                              hipStream_t stream) {
    (void)in_sizes; (void)n_in; (void)out_size; (void)ws_size;
    constexpr int Bc = 4, Tc = 4096, DIMc = 1024, Hc = 16, KP = 256, DHc = 64;

    const float* x    = (const float*)d_in[0];
    const float* proj = (const float*)d_in[1];
    const float* Wqkv = (const float*)d_in[2];
    const float* W0   = (const float*)d_in[3];
    float* out = (float*)d_out;

    const size_t nX    = (size_t)Bc * Tc * DIMc;        // 16,777,216
    const size_t nWqkv = (size_t)DIMc * 3 * DIMc;       //  3,145,728
    const size_t nW0   = (size_t)DIMc * DIMc;
    const size_t nProj = (size_t)Tc * KP;
    const size_t nQKV  = (size_t)Bc * Tc * 3 * DIMc;    // 50,331,648
    const size_t nKP   = (size_t)Bc * Hc * KP * DHc;    //  1,048,576

    char* p = (char*)d_ws;
    auto alloc = [&](size_t elems) -> __bf16* {
        __bf16* r = (__bf16*)p;
        p += ((elems * sizeof(__bf16)) + 255) & ~(size_t)255;
        return r;
    };
    __bf16* x_bf    = alloc(nX);
    __bf16* wqkv_bf = alloc(nWqkv);
    __bf16* w0_bf   = alloc(nW0);
    __bf16* projT   = alloc(nProj);    // [KPROJ][T]
    __bf16* qkvb    = alloc(nQKV);     // [B*T][3072], f = d*48 + k*16 + h
    __bf16* kpb     = alloc(nKP);      // [B*H][256][64]
    __bf16* vpb     = alloc(nKP);
    __bf16* aout    = alloc(nX);       // [B*T][1024]

    dim3 cb(256);
    f32_to_bf16_kernel<<<2048, cb, 0, stream>>>(x, x_bf, nX);
    f32_to_bf16_kernel<<<1024, cb, 0, stream>>>(Wqkv, wqkv_bf, nWqkv);
    f32_to_bf16_kernel<<<512,  cb, 0, stream>>>(W0, w0_bf, nW0);
    transpose_f32_to_bf16_kernel<<<1024, cb, 0, stream>>>(proj, projT, Tc, KP);

    // GEMM1: qkv = x @ Wqkv   [16384x1024]x[1024x3072]  (fast path)
    {
        dim3 g(3072 / 128, 16384 / 128, 1);
        gemm_bf16_wmma_fast<true><<<g, 256, 0, stream>>>(
            x_bf, wqkv_bf, qkvb, 16384, 3072, 1024);
    }
    // GEMM2/3: kp,vp = projT @ {k,v}-view   batched over z = b*16+h
    {
        dim3 g(1, 2, Bc * Hc);
        const long long bB0 = (long long)Tc * 3072;              // per-b stride in qkv
        const long long bC0 = (long long)Hc * KP * DHc;          // per-b stride in kp
        const long long bC1 = (long long)KP * DHc;               // per-h stride in kp
        gemm_bf16_wmma<true><<<g, 256, 0, stream>>>(
            projT, qkvb + 16, kpb, KP, DHc, Tc,
            (long long)Tc, 1LL, 3072LL, 48LL, (long long)DHc, 1LL,
            Hc, 0LL, 0LL, bB0, 1LL, bC0, bC1);
        gemm_bf16_wmma<true><<<g, 256, 0, stream>>>(
            projT, qkvb + 32, vpb, KP, DHc, Tc,
            (long long)Tc, 1LL, 3072LL, 48LL, (long long)DHc, 1LL,
            Hc, 0LL, 0LL, bB0, 1LL, bC0, bC1);
    }
    // Fused attention
    {
        dim3 g(Tc / 128, Bc * Hc);
        linformer_attn_kernel<<<g, 256, 0, stream>>>(qkvb, kpb, vpb, aout);
    }
    // Final: out = aout @ W0   [16384x1024]x[1024x1024] -> fp32  (fast path)
    {
        dim3 g(1024 / 128, 16384 / 128, 1);
        gemm_bf16_wmma_fast<false><<<g, 256, 0, stream>>>(
            aout, w0_bf, out, 16384, 1024, 1024);
    }
}
